// SioConvLayer_42167988912904
// MI455X (gfx1250) — matrix operand order repaired and checked by reading, not compiled, not run on hardware
//
#include <hip/hip_runtime.h>
#include <math.h>

#define BB 2
#define LL 1024
#define DD 1024
#define HH 8
#define II 64
#define HI6 3072
#define HI2 2048
#define NTOK (BB*LL)

typedef __attribute__((ext_vector_type(16))) __bf16 v16bf;
typedef __attribute__((ext_vector_type(8)))  __bf16 v8bf;
typedef __attribute__((ext_vector_type(8)))  float  v8f;

__device__ __forceinline__ unsigned short f2bf(float f) {
    union { float f; unsigned u; } c; c.f = f;
    unsigned r = c.u + 0x7FFFu + ((c.u >> 16) & 1u);
    return (unsigned short)(r >> 16);
}

__device__ __forceinline__ v8f wmma_acc(v16bf a, v16bf b, v8f c) {
    return __builtin_amdgcn_wmma_f32_16x16x32_bf16(false, a, false, b, (short)0, c, false, false);
}

// A-fragment (16x32 bf16): lane ln -> row = ln%16; element e -> K = 16*(e/8) + (ln/16)*8 + e%8
__device__ __forceinline__ v16bf load_a_frag(const unsigned short* base, int row, int ldk, int k0, int hi) {
    union { v16bf v; v8bf h[2]; } a;
    a.h[0] = *(const v8bf*)(base + (size_t)row * ldk + k0 + hi * 8);
    a.h[1] = *(const v8bf*)(base + (size_t)row * ldk + k0 + 16 + hi * 8);
    return a.v;
}
// B-fragment (32x16 bf16, B given as [N][K] row-major): lane ln -> col = ln%16; element e -> K = 16*(e/8) + (ln/16)*8 + e%8
__device__ __forceinline__ v16bf load_b_frag(const unsigned short* base, int col, int ldk, int k0, int hi) {
    union { v16bf v; v8bf h[2]; } b;
    b.h[0] = *(const v8bf*)(base + (size_t)col * ldk + k0 + hi * 8);
    b.h[1] = *(const v8bf*)(base + (size_t)col * ldk + k0 + 16 + hi * 8);
    return b.v;
}

// ---------------- convert f32 -> bf16 ----------------
__global__ void k_f2bf(const float* __restrict__ src, unsigned short* __restrict__ dst, int n) {
    int i = blockIdx.x * 256 + threadIdx.x;
    if (i < n) dst[i] = f2bf(src[i]);
}

// ---------------- generic WMMA GEMM: C[M,N] = A[M,K] * B[N,K]^T + bias ----------------
__global__ void k_gemm_bf16(const unsigned short* __restrict__ A,
                            const unsigned short* __restrict__ Bm,
                            const float* __restrict__ bias,
                            float* __restrict__ C, int M, int N, int K) {
    int wave = (blockIdx.x * blockDim.x + threadIdx.x) >> 5;
    int lane = threadIdx.x & 31;
    int tilesN = N >> 4;
    int tm = wave / tilesN, tn = wave % tilesN;
    int l15 = lane & 15, hi = lane >> 4;
    const unsigned short* arow = A + (size_t)(tm * 16 + l15) * K;
    const unsigned short* brow = Bm + (size_t)(tn * 16 + l15) * K;
    v8f acc = {};
    for (int k0 = 0; k0 < K; k0 += 32) {
        __builtin_prefetch(arow + k0 + 256, 0, 0);
        __builtin_prefetch(brow + k0 + 256, 0, 0);
        union { v16bf v; v8bf h[2]; } a;
        a.h[0] = *(const v8bf*)(arow + k0 + hi * 8);
        a.h[1] = *(const v8bf*)(arow + k0 + 16 + hi * 8);
        union { v16bf v; v8bf h[2]; } bu;
        bu.h[0] = *(const v8bf*)(brow + k0 + hi * 8);
        bu.h[1] = *(const v8bf*)(brow + k0 + 16 + hi * 8);
        v16bf b = bu.v;
        acc = wmma_acc(a.v, b, acc);
    }
    int col = tn * 16 + l15;
    float bv = bias ? bias[col] : 0.0f;
#pragma unroll
    for (int r = 0; r < 8; ++r) {
        int row = tm * 16 + r + 8 * hi;
        C[(size_t)row * N + col] = acc[r] + bv;
    }
}

// ---------------- small projections: a_angle / ln_a_abs / scale_qk  [B,L,H] ----------------
__global__ void k_small_proj(const float* __restrict__ x,
                             const float* __restrict__ Wa, const float* __restrict__ ba,
                             const float* __restrict__ Wl, const float* __restrict__ bl,
                             const float* __restrict__ Wp, const float* __restrict__ bp,
                             float* __restrict__ aang, float* __restrict__ lnabs,
                             float* __restrict__ pscale) {
    int tok = blockIdx.x;           // 0..NTOK-1
    int o = blockIdx.y;             // 0..23
    int h = o & 7;
    const float* W = (o < 8) ? Wa : (o < 16) ? Wl : Wp;
    const float* xrow = x + (size_t)tok * DD;
    const float* wrow = W + (size_t)h * DD;
    float s = 0.f;
    for (int d = threadIdx.x; d < DD; d += 64) s += xrow[d] * wrow[d];
    __shared__ float red[64];
    red[threadIdx.x] = s;
    __syncthreads();
    if (threadIdx.x == 0) {
        float t = 0.f;
        for (int i = 0; i < 64; ++i) t += red[i];
        if (o < 8)        aang[tok * HH + h]   = tanhf(t + ba[h]);
        else if (o < 16)  lnabs[tok * HH + h]  = -expf(t + bl[h]);
        else              pscale[tok * HH + h] = -1.0f / (1.0f + expf(-(t + bp[h])));
    }
}

// ---------------- complex inclusive scan of ln_a along L -> A  [B,L,H] ----------------
__global__ void k_scan(const float* __restrict__ lnabs, const float* __restrict__ aang,
                       float* __restrict__ Ar, float* __restrict__ Ai) {
    int b = blockIdx.x >> 3, h = blockIdx.x & 7;
    int l = threadIdx.x;            // 1024 threads
    float p0 = powf(1e-4f, (float)(h * II) / (float)(HH * II));
    float vr = lnabs[(b * LL + l) * HH + h] * p0;
    float vi = aang [(b * LL + l) * HH + h] * p0;
    int lane = l & 31, w = l >> 5;
    for (int off = 1; off < 32; off <<= 1) {
        float tr = __shfl_up(vr, off, 32), ti = __shfl_up(vi, off, 32);
        if (lane >= off) { vr += tr; vi += ti; }
    }
    __shared__ float sr[32], si[32];
    if (lane == 31) { sr[w] = vr; si[w] = vi; }
    __syncthreads();
    if (w == 0) {
        float xr = sr[lane], xi = si[lane];
        for (int off = 1; off < 32; off <<= 1) {
            float tr = __shfl_up(xr, off, 32), ti = __shfl_up(xi, off, 32);
            if (lane >= off) { xr += tr; xi += ti; }
        }
        sr[lane] = xr; si[lane] = xi;
    }
    __syncthreads();
    if (w > 0) { vr += sr[w - 1]; vi += si[w - 1]; }
    Ar[(b * LL + l) * HH + h] = vr;
    Ai[(b * LL + l) * HH + h] = vi;
}

// ---------------- qkv nonlinearity + phase rotation -> qp,kp,v  [B,H,L,I] ----------------
__global__ void k_qkv_phase(const float* __restrict__ t, const float* __restrict__ pscale,
                            float* __restrict__ qpr, float* __restrict__ qpi,
                            float* __restrict__ kpr, float* __restrict__ kpi,
                            float* __restrict__ vre, float* __restrict__ vim) {
    int idx = blockIdx.x * 256 + threadIdx.x;        // over B*L*H*I
    int i = idx & 63;
    int h = (idx >> 6) & 7;
    int l = (idx >> 9) & (LL - 1);
    int b = idx >> 19;
    int tok = b * LL + l;
    const float* tr = t + (size_t)tok * HI6 + (size_t)(h * II + i) * 6;
    float qr = tr[0], qi = tr[1], kr = tr[2], ki = tr[3], vr = tr[4], vi = tr[5];
    float nq = 1.0f / (1.0f + sqrtf(qr * qr + qi * qi));
    float nk = 1.0f / (1.0f + sqrtf(kr * kr + ki * ki));
    float nv = 1.0f / (1.0f + sqrtf(vr * vr + vi * vi));
    qr *= nq; qi *= nq; kr *= nk; ki *= nk; vr *= nv; vi *= nv;

    float pang = powf(1e-4f, (float)(h * II + i) / (float)(HH * II));
    float psc  = expf(logf(32.0f) + (logf(1024.0f) - logf(32.0f)) * (float)h / (float)(HH - 1));
    float diff = pscale[tok * HH + h] * psc * pang;
    float dq = ((i & 1) == 0) ? diff : 0.0f;
    float dk = ((i & 1) == 1) ? diff : 0.0f;
    float phase = pang * (float)l;
    float aq = phase + dq, ak = -phase + dk;
    float cq = cosf(aq), sq = sinf(aq);
    float ck = cosf(ak), sk = sinf(ak);
    size_t o = (((size_t)(b * HH + h)) * LL + l) * II + i;
    qpr[o] = qr * cq - qi * sq;  qpi[o] = qr * sq + qi * cq;
    kpr[o] = kr * ck - ki * sk;  kpi[o] = kr * sk + ki * ck;
    vre[o] = vr;                 vim[o] = vi;
}

// ---------------- chunked complex linear attention (WMMA core) ----------------
// grid = B*H blocks, block = 512 (16 waves). One (b,h) per block, sequential over 16 chunks of 64.
__global__ void k_attention(const float* __restrict__ qpr, const float* __restrict__ qpi,
                            const float* __restrict__ kpr, const float* __restrict__ kpi,
                            const float* __restrict__ vre, const float* __restrict__ vim,
                            const float* __restrict__ Ar,  const float* __restrict__ Ai,
                            const float* __restrict__ hid_r, const float* __restrict__ hid_i,
                            float* __restrict__ hr_out, float* __restrict__ hi_out,
                            float* __restrict__ hid_out) {
    int b = blockIdx.x >> 3, h = blockIdx.x & 7;
    int tid = threadIdx.x;
    extern __shared__ char smem[];
    unsigned short* Qr   = (unsigned short*)smem;
    unsigned short* Qi   = Qr   + 4096;
    unsigned short* QiN  = Qi   + 4096;
    unsigned short* Kr   = QiN  + 4096;
    unsigned short* Ki   = Kr   + 4096;
    unsigned short* KTr  = Ki   + 4096;
    unsigned short* KTi  = KTr  + 4096;
    unsigned short* KTiN = KTi  + 4096;
    unsigned short* VTr  = KTiN + 4096;
    unsigned short* VTi  = VTr  + 4096;
    unsigned short* Gr   = VTi  + 4096;
    unsigned short* Gi   = Gr   + 4096;
    unsigned short* GiN  = Gi   + 4096;
    unsigned short* STr  = GiN  + 4096;
    unsigned short* STi  = STr  + 4096;
    float* Sr   = (float*)(STi + 4096);
    float* Si   = Sr + 4096;
    float* wend = Si + 4096;    // 2 floats

    // ---- init state S0[i][j] = hidden[b,h,i,j] * exp(1j * p_angle[h,i]) ----
    for (int e = tid; e < 4096; e += 512) {
        int i = e >> 6, j = e & 63;
        size_t ho = (((size_t)(b * HH + h)) * II + i) * II + j;
        float hr0 = hid_r[ho], hi0 = hid_i[ho];
        float pa = powf(1e-4f, (float)(h * II + i) / (float)(HH * II));
        float cc = cosf(pa), ss = sinf(pa);
        Sr[i * 64 + j] = hr0 * cc - hi0 * ss;
        Si[i * 64 + j] = hr0 * ss + hi0 * cc;
    }
    __syncthreads();
    for (int e = tid; e < 4096; e += 512) {
        int j = e >> 6, i = e & 63;
        STr[j * 64 + i] = f2bf(Sr[i * 64 + j]);
        STi[j * 64 + i] = f2bf(Si[i * 64 + j]);
    }
    __syncthreads();

    int wv = tid >> 5, ln = tid & 31;
    int l15 = ln & 15, hi2 = ln >> 4;
    int lt = wv >> 2, mt = wv & 3;

    for (int c = 0; c < LL / 64; ++c) {
        // ---- stage Q~, K~, KT, VT (bf16) with decay factors folded in ----
        {
            int sl = tid >> 3;              // 0..63 local row
            int si0 = (tid & 7) * 8;
            int gl = c * 64 + sl;
            float ArL = Ar[(b * LL + gl) * HH + h];
            float AiL = Ai[(b * LL + gl) * HH + h];
            float Arb = 0.f, Aib = 0.f;
            if (c > 0) {
                int pb = c * 64 - 1;
                Arb = Ar[(b * LL + pb) * HH + h];
                Aib = Ai[(b * LL + pb) * HH + h];
            }
            float ang = AiL - Aib;
            float ew = expf(ArL - Arb), eu = expf(Arb - ArL);
            float cA = cosf(ang), sA = sinf(ang);
            float wr = ew * cA, wi = ew * sA;
            float ur = eu * cA, ui = -eu * sA;
            size_t base = (((size_t)(b * HH + h)) * LL + gl) * II;
#pragma unroll
            for (int t2 = 0; t2 < 8; ++t2) {
                int i = si0 + t2;
                float qr = qpr[base + i], qi = qpi[base + i];
                float Qtr = qr * wr - qi * wi, Qti = qr * wi + qi * wr;
                Qr [sl * 64 + i] = f2bf(Qtr);
                Qi [sl * 64 + i] = f2bf(Qti);
                QiN[sl * 64 + i] = f2bf(-Qti);
                float kr = kpr[base + i], ki = kpi[base + i];
                float Ktr = kr * ur - ki * ui, Kti = kr * ui + ki * ur;
                Kr  [sl * 64 + i] = f2bf(Ktr);
                Ki  [sl * 64 + i] = f2bf(Kti);
                KTr [i * 64 + sl] = f2bf(Ktr);
                KTi [i * 64 + sl] = f2bf(Kti);
                KTiN[i * 64 + sl] = f2bf(-Kti);
                VTr [i * 64 + sl] = f2bf(vre[base + i]);
                VTi [i * 64 + sl] = f2bf(vim[base + i]);
            }
            if (tid == 0) {
                int le = c * 64 + 63;
                float Are = Ar[(b * LL + le) * HH + h], Aie = Ai[(b * LL + le) * HH + h];
                float e2 = expf(Are - Arb), a2 = Aie - Aib;
                wend[0] = e2 * cosf(a2);
                wend[1] = e2 * sinf(a2);
            }
        }
        __syncthreads();

        // ---- G = mask(Q~ K~^T)  (complex), tiles (lt,mt) ----
        {
            v8f gr = {}, gi = {};
            if (lt >= mt) {
#pragma unroll
                for (int k0 = 0; k0 < 64; k0 += 32) {
                    v16bf aqr  = load_a_frag(Qr,  lt * 16 + l15, 64, k0, hi2);
                    v16bf aqi  = load_a_frag(Qi,  lt * 16 + l15, 64, k0, hi2);
                    v16bf aqiN = load_a_frag(QiN, lt * 16 + l15, 64, k0, hi2);
                    v16bf bkr  = load_b_frag(Kr,  mt * 16 + l15, 64, k0, hi2);
                    v16bf bki  = load_b_frag(Ki,  mt * 16 + l15, 64, k0, hi2);
                    gr = wmma_acc(aqr,  bkr, gr);
                    gr = wmma_acc(aqiN, bki, gr);
                    gi = wmma_acc(aqr,  bki, gi);
                    gi = wmma_acc(aqi,  bkr, gi);
                }
            }
#pragma unroll
            for (int r = 0; r < 8; ++r) {
                int row = lt * 16 + r + 8 * hi2, col = mt * 16 + l15;
                bool keep = (lt > mt) || ((lt == mt) && (row >= col));
                float grv = keep ? gr[r] : 0.f, giv = keep ? gi[r] : 0.f;
                Gr [row * 64 + col] = f2bf(grv);
                Gi [row * 64 + col] = f2bf(giv);
                GiN[row * 64 + col] = f2bf(-giv);
            }
        }
        __syncthreads();

        // ---- H = G V + Q~ S_prev ; DeltaS = K~^T V ; S = wend*(S + DeltaS) ----
        {
            int jt = mt;
            v8f hrA = {}, hiA = {};
#pragma unroll
            for (int k0 = 0; k0 < 64; k0 += 32) {
                v16bf agr  = load_a_frag(Gr,  lt * 16 + l15, 64, k0, hi2);
                v16bf agi  = load_a_frag(Gi,  lt * 16 + l15, 64, k0, hi2);
                v16bf agiN = load_a_frag(GiN, lt * 16 + l15, 64, k0, hi2);
                v16bf bvr  = load_b_frag(VTr, jt * 16 + l15, 64, k0, hi2);
                v16bf bvi  = load_b_frag(VTi, jt * 16 + l15, 64, k0, hi2);
                hrA = wmma_acc(agr,  bvr, hrA);
                hrA = wmma_acc(agiN, bvi, hrA);
                hiA = wmma_acc(agr,  bvi, hiA);
                hiA = wmma_acc(agi,  bvr, hiA);
                v16bf aqr  = load_a_frag(Qr,  lt * 16 + l15, 64, k0, hi2);
                v16bf aqi  = load_a_frag(Qi,  lt * 16 + l15, 64, k0, hi2);
                v16bf aqiN = load_a_frag(QiN, lt * 16 + l15, 64, k0, hi2);
                v16bf bsr  = load_b_frag(STr, jt * 16 + l15, 64, k0, hi2);
                v16bf bsi  = load_b_frag(STi, jt * 16 + l15, 64, k0, hi2);
                hrA = wmma_acc(aqr,  bsr, hrA);
                hrA = wmma_acc(aqiN, bsi, hrA);
                hiA = wmma_acc(aqr,  bsi, hiA);
                hiA = wmma_acc(aqi,  bsr, hiA);
            }
#pragma unroll
            for (int r = 0; r < 8; ++r) {
                int gl2 = c * 64 + lt * 16 + r + 8 * hi2;
                int j = jt * 16 + l15;
                size_t o = (((size_t)(b * LL + gl2)) * HH + h) * II + j;
                hr_out[o] = hrA[r];
                hi_out[o] = hiA[r];
            }
            v8f dsr = {}, dsi = {};
#pragma unroll
            for (int k0 = 0; k0 < 64; k0 += 32) {
                v16bf aktr  = load_a_frag(KTr,  lt * 16 + l15, 64, k0, hi2);
                v16bf akti  = load_a_frag(KTi,  lt * 16 + l15, 64, k0, hi2);
                v16bf aktiN = load_a_frag(KTiN, lt * 16 + l15, 64, k0, hi2);
                v16bf bvr   = load_b_frag(VTr,  jt * 16 + l15, 64, k0, hi2);
                v16bf bvi   = load_b_frag(VTi,  jt * 16 + l15, 64, k0, hi2);
                dsr = wmma_acc(aktr,  bvr, dsr);
                dsr = wmma_acc(aktiN, bvi, dsr);
                dsi = wmma_acc(aktr,  bvi, dsi);
                dsi = wmma_acc(akti,  bvr, dsi);
            }
            float wer = wend[0], wei = wend[1];
#pragma unroll
            for (int r = 0; r < 8; ++r) {
                int i_ = lt * 16 + r + 8 * hi2, j_ = jt * 16 + l15;
                float nr = Sr[i_ * 64 + j_] + dsr[r];
                float ni = Si[i_ * 64 + j_] + dsi[r];
                Sr[i_ * 64 + j_] = wer * nr - wei * ni;
                Si[i_ * 64 + j_] = wer * ni + wei * nr;
            }
        }
        __syncthreads();
        // ---- restage ST (bf16 transpose of S) for next chunk ----
        for (int e = tid; e < 4096; e += 512) {
            int j = e >> 6, i = e & 63;
            STr[j * 64 + i] = f2bf(Sr[i * 64 + j]);
            STi[j * 64 + i] = f2bf(Si[i * 64 + j]);
        }
        __syncthreads();
    }

    // ---- hidden_next = P^(L-1)[i] * S_final, interleaved re/im ----
    for (int e = tid; e < 4096; e += 512) {
        int i = e >> 6, j = e & 63;
        float pa = powf(1e-4f, (float)(h * II + i) / (float)(HH * II)) * (float)(LL - 1);
        float cc = cosf(pa), ss = sinf(pa);
        float sr0 = Sr[i * 64 + j], si0 = Si[i * 64 + j];
        size_t o = ((((size_t)(b * HH + h)) * II + i) * II + j) * 2;
        hid_out[o + 0] = cc * sr0 - ss * si0;
        hid_out[o + 1] = cc * si0 + ss * sr0;
    }
}

// ---------------- GroupNorm + SiLU gate -> z (bf16, [NTOK, 2048]) ----------------
__global__ void k_gn_gate(const float* __restrict__ hr, const float* __restrict__ hi,
                          const float* __restrict__ g,
                          const float* __restrict__ gn_w, const float* __restrict__ gn_b,
                          unsigned short* __restrict__ z) {
    int tok = blockIdx.x;
    int wv = threadIdx.x >> 5;      // group h (8 waves)
    int ln = threadIdx.x & 31;
    const float* hrp = hr + ((size_t)tok * HH + wv) * II;
    const float* hip = hi + ((size_t)tok * HH + wv) * II;
    float v4[4], sum = 0.f, sq = 0.f;
#pragma unroll
    for (int t = 0; t < 4; ++t) {
        int cch = ln + 32 * t;              // 0..127
        int i = cch >> 1;
        float val = (cch & 1) ? hip[i] : hrp[i];
        v4[t] = val; sum += val; sq += val * val;
    }
#pragma unroll
    for (int off = 16; off >= 1; off >>= 1) {
        sum += __shfl_xor(sum, off, 32);
        sq  += __shfl_xor(sq,  off, 32);
    }
    float mean = sum * (1.0f / 128.0f);
    float var = sq * (1.0f / 128.0f) - mean * mean;
    float rstd = rsqrtf(var + 1e-5f);
    float gw = gn_w[wv], gb = gn_b[wv];
#pragma unroll
    for (int t = 0; t < 4; ++t) {
        int cch = ln + 32 * t;
        int ch = wv * 128 + cch;
        float hn = (v4[t] - mean) * rstd * gw + gb;
        float gg = g[(size_t)tok * HI2 + ch];
        float zl = hn * (gg / (1.0f + expf(-gg)));
        z[(size_t)tok * HI2 + ch] = f2bf(zl);
    }
}

extern "C" void kernel_launch(void* const* d_in, const int* in_sizes, int n_in,
                              void* d_out, int out_size, void* d_ws, size_t ws_size,
                              hipStream_t stream) {
    const float* x      = (const float*)d_in[0];
    const float* hid_r  = (const float*)d_in[1];
    const float* hid_i  = (const float*)d_in[2];
    const float* W_qkv  = (const float*)d_in[3];
    const float* b_qkv  = (const float*)d_in[4];
    const float* W_g    = (const float*)d_in[5];
    const float* b_g    = (const float*)d_in[6];
    const float* W_aang = (const float*)d_in[7];
    const float* b_aang = (const float*)d_in[8];
    const float* W_lnab = (const float*)d_in[9];
    const float* b_lnab = (const float*)d_in[10];
    const float* W_psc  = (const float*)d_in[11];
    const float* b_psc  = (const float*)d_in[12];
    const float* W_y    = (const float*)d_in[13];
    const float* b_y    = (const float*)d_in[14];
    const float* gn_w   = (const float*)d_in[15];
    const float* gn_b   = (const float*)d_in[16];

    float* y_out = (float*)d_out;                       // [NTOK, D]
    float* hid_out = (float*)d_out + (size_t)NTOK * DD; // [B,H,I,I] complex interleaved

    char* w = (char*)d_ws;
    auto carve = [&](size_t bytes) -> char* {
        char* p = w; w += (bytes + 255) & ~(size_t)255; return p;
    };
    unsigned short* xb    = (unsigned short*)carve((size_t)NTOK * DD * 2);
    unsigned short* wqkvb = (unsigned short*)carve((size_t)HI6 * DD * 2);
    unsigned short* wgb   = (unsigned short*)carve((size_t)HI2 * DD * 2);
    unsigned short* wyb   = (unsigned short*)carve((size_t)DD * HI2 * 2);
    float* t_buf  = (float*)carve((size_t)NTOK * HI6 * 4);
    float* g_buf  = (float*)carve((size_t)NTOK * HI2 * 4);
    float* aang   = (float*)carve((size_t)NTOK * HH * 4);
    float* lnabs  = (float*)carve((size_t)NTOK * HH * 4);
    float* pscale = (float*)carve((size_t)NTOK * HH * 4);
    float* Ar     = (float*)carve((size_t)NTOK * HH * 4);
    float* Ai     = (float*)carve((size_t)NTOK * HH * 4);
    size_t nqk = (size_t)BB * HH * LL * II;
    float* qpr = (float*)carve(nqk * 4);
    float* qpi = (float*)carve(nqk * 4);
    float* kpr = (float*)carve(nqk * 4);
    float* kpi = (float*)carve(nqk * 4);
    float* vre = (float*)carve(nqk * 4);
    float* vim = (float*)carve(nqk * 4);
    float* hr_buf = (float*)carve((size_t)NTOK * HH * II * 4);
    float* hi_buf = (float*)carve((size_t)NTOK * HH * II * 4);
    unsigned short* z_buf = (unsigned short*)carve((size_t)NTOK * HI2 * 2);

    // 1) f32 -> bf16 conversions
    k_f2bf<<<(NTOK * DD + 255) / 256, 256, 0, stream>>>(x, xb, NTOK * DD);
    k_f2bf<<<(HI6 * DD + 255) / 256, 256, 0, stream>>>(W_qkv, wqkvb, HI6 * DD);
    k_f2bf<<<(HI2 * DD + 255) / 256, 256, 0, stream>>>(W_g, wgb, HI2 * DD);
    k_f2bf<<<(DD * HI2 + 255) / 256, 256, 0, stream>>>(W_y, wyb, DD * HI2);

    // 2) small projections + scan
    k_small_proj<<<dim3(NTOK, 24), 64, 0, stream>>>(x, W_aang, b_aang, W_lnab, b_lnab,
                                                    W_psc, b_psc, aang, lnabs, pscale);
    k_scan<<<BB * HH, LL, 0, stream>>>(lnabs, aang, Ar, Ai);

    // 3) big projections (WMMA)
    k_gemm_bf16<<<(NTOK / 16) * (HI6 / 16) / 8, 256, 0, stream>>>(xb, wqkvb, b_qkv, t_buf,
                                                                  NTOK, HI6, DD);
    k_gemm_bf16<<<(NTOK / 16) * (HI2 / 16) / 8, 256, 0, stream>>>(xb, wgb, b_g, g_buf,
                                                                  NTOK, HI2, DD);

    // 4) qkv nonlinearity + phase
    k_qkv_phase<<<(BB * LL * HH * II) / 256, 256, 0, stream>>>(t_buf, pscale,
                                                               qpr, qpi, kpr, kpi, vre, vim);

    // 5) chunked complex attention (WMMA) + hidden_next
    size_t att_smem = 15 * 4096 * sizeof(unsigned short) + 2 * 4096 * sizeof(float) + 256;
    k_attention<<<BB * HH, 512, att_smem, stream>>>(qpr, qpi, kpr, kpi, vre, vim, Ar, Ai,
                                                    hid_r, hid_i, hr_buf, hi_buf, hid_out);

    // 6) GroupNorm + SiLU gate
    k_gn_gate<<<NTOK, 256, 0, stream>>>(hr_buf, hi_buf, g_buf, gn_w, gn_b, z_buf);

    // 7) output projection (WMMA)
    k_gemm_bf16<<<(NTOK / 16) * (DD / 16) / 8, 256, 0, stream>>>(z_buf, wyb, b_y, y_out,
                                                                 NTOK, DD, HI2);
}
